// Temporal_Active_Focus_corr_74517682585712
// MI455X (gfx1250) — matrix operand order repaired and checked by reading, not compiled, not run on hardware
//
#include <hip/hip_runtime.h>

// ---------------------------------------------------------------------------
// Temporal_Active_Focus_corr for MI455X (gfx1250, wave32, WMMA 16x16x32 f16,
// async global->LDS loads, packed weight fragments, vectorized LDS fragments)
// ---------------------------------------------------------------------------

typedef __attribute__((ext_vector_type(16))) _Float16 v16h;
typedef __attribute__((ext_vector_type(8)))  float    v8f;

#define HPIX 128
#define WPIX 160
#define HW   (HPIX * WPIX)          // 20480
#define CE   128
#define NHW  (CE * HW)              // 2621440
#define H0   256
#define W0   320

__constant__ float c_deltas[4] = {0.0f, 5.0f, 10.0f, 25.0f};

// ---------------------------------------------------------------------------
// Async global -> LDS copy (16 B per call), with portable fallback
// ---------------------------------------------------------------------------
#if defined(__has_builtin)
#if __has_builtin(__builtin_amdgcn_global_load_async_to_lds_b128)
#define HAVE_ASYNC_LDS 1
#endif
#endif

typedef int vi4 __attribute__((vector_size(16)));
typedef __attribute__((address_space(1))) vi4 gvi4;   // global int4 vector
typedef __attribute__((address_space(3))) vi4 lvi4;   // LDS int4 vector

__device__ __forceinline__ void copy16_g2l(const _Float16* g, _Float16* l) {
#ifdef HAVE_ASYNC_LDS
  __builtin_amdgcn_global_load_async_to_lds_b128(
      (gvi4*)(uintptr_t)g,
      (lvi4*)(unsigned)(uintptr_t)l,
      0, 0);
#else
  *(uint4*)l = *(const uint4*)g;
#endif
}

__device__ __forceinline__ void async_copies_wait() {
#ifdef HAVE_ASYNC_LDS
  asm volatile("s_wait_asynccnt 0x0" ::: "memory");
#endif
}

// ---------------------------------------------------------------------------
// WMMA fragment helpers (wave32 layouts per CDNA5 ISA 7.12.2)
// A fragment per lane = two contiguous 8-half runs: k=kb..kb+7 and k=16+kb..+7.
// B/K^T fragment per lane = same two 8-half runs as A: k=kb..kb+7 and k=16+kb..+7.
// ---------------------------------------------------------------------------

// load A fragment runs (base already includes row*stride + k0 + kb)
__device__ __forceinline__ v16h ld_a_fast(const _Float16* base) {
  v16h a;
  ((uint2*)&a)[0] = *(const uint2*)(base + 0);
  ((uint2*)&a)[1] = *(const uint2*)(base + 4);
  ((uint2*)&a)[2] = *(const uint2*)(base + 16);
  ((uint2*)&a)[3] = *(const uint2*)(base + 20);
  return a;
}

// load 16 contiguous halves
__device__ __forceinline__ v16h ld_c16_fast(const _Float16* base) {
  v16h b;
  ((uint2*)&b)[0] = *(const uint2*)(base + 0);
  ((uint2*)&b)[1] = *(const uint2*)(base + 4);
  ((uint2*)&b)[2] = *(const uint2*)(base + 8);
  ((uint2*)&b)[3] = *(const uint2*)(base + 12);
  return b;
}

// zero A-fragment elements with tile-local k >= KMAX (compile-time pattern)
template <int KMAX>
__device__ __forceinline__ v16h mask_a(v16h a, int kb) {
  if constexpr (KMAX < 32) {
#pragma unroll
    for (int j = 0; j < 8; ++j) {
      a[j]     = (kb + j      < KMAX) ? a[j]     : (_Float16)0.f;
      a[8 + j] = (16 + kb + j < KMAX) ? a[8 + j] : (_Float16)0.f;
    }
  }
  return a;
}

// zero B-fragment elements with tile-local k >= KMAX
template <int KMAX>
__device__ __forceinline__ v16h mask_c16(v16h b, int kb16) {
  if constexpr (KMAX < 32) {
#pragma unroll
    for (int j = 0; j < 16; ++j)
      b[j] = (kb16 + j < KMAX) ? b[j] : (_Float16)0.f;
  }
  return b;
}

// packed weight B fragment: 32 B contiguous per lane, pre-padded at pack time
__device__ __forceinline__ v16h ld_frag_b_pk(const _Float16* pw, int tile) {
  int lane = threadIdx.x & 31;
  const _Float16* p = pw + ((size_t)tile * 32 + lane) * 16;
  v16h b;
  ((uint4*)&b)[0] = *(const uint4*)(p + 0);
  ((uint4*)&b)[1] = *(const uint4*)(p + 8);
  return b;
}

// D/C (16x16 f32): VGPR r -> row m0 + r + (lane>=16 ? 8 : 0); col n0 + lane&15
__device__ __forceinline__ void st_frag_d_f16(_Float16* dst, int stride, int m0,
                                              int n0, int nmax, v8f d) {
  int lane = threadIdx.x & 31;
  int n = lane & 15;
  if (n < nmax) {
#pragma unroll
    for (int r = 0; r < 8; ++r) {
      int m = m0 + r + ((lane >> 4) << 3);
      dst[m * stride + n0 + n] = (_Float16)d[r];
    }
  }
}

// transposed store for V: Vt[(n0+n)*128 + m]
__device__ __forceinline__ void st_frag_d_f16_T(_Float16* dst, int m0, int n0, v8f d) {
  int lane = threadIdx.x & 31;
  int n = lane & 15;
#pragma unroll
  for (int r = 0; r < 8; ++r) {
    int m = m0 + r + ((lane >> 4) << 3);
    dst[(n0 + n) * 128 + m] = (_Float16)d[r];
  }
}

__device__ __forceinline__ void st_frag_d_f32(float* dst, int stride, int m0, int n0,
                                              v8f d) {
  int lane = threadIdx.x & 31;
  int n = n0 + (lane & 15);
#pragma unroll
  for (int r = 0; r < 8; ++r) {
    int m = m0 + r + ((lane >> 4) << 3);
    dst[m * stride + n] = d[r];
  }
}

#define WMMA_F16(a, b, c) \
  __builtin_amdgcn_wmma_f32_16x16x32_f16(false, (a), false, (b), (short)0, (c), false, false)

// ---------------------------------------------------------------------------
// Weight packing: C x C f32 row-major -> WMMA B-fragment order, K padded to CKp
// layout: [tile = ktile*(C/16)+ntile][lane 0..31][j 0..15]
// ---------------------------------------------------------------------------
__global__ void pack_w_kernel(const float* __restrict__ w, _Float16* __restrict__ dst,
                              int C, int CKp) {
  int NTl = C / 16;
  int total = (CKp / 32) * NTl * 32 * 16;
  int idx = blockIdx.x * 256 + threadIdx.x;
  if (idx >= total) return;
  int j = idx & 15;
  int lane = (idx >> 4) & 31;
  int tile = idx >> 9;
  int ntile = tile % NTl;
  int ktile = tile / NTl;
  int n = ntile * 16 + (lane & 15);
  int k = ktile * 32 + ((lane >> 4) & 1) * 8 + (j & 7) + (j >> 3) * 16;
  dst[idx] = (k < C) ? (_Float16)w[k * C + n] : (_Float16)0.f;
}

// ---------------------------------------------------------------------------
// Elementwise kernels
// ---------------------------------------------------------------------------

// ref transform: x (16,H0,W0) -> refin (4,16,H0,W0)
__global__ void ref_transform_kernel(const float* __restrict__ x,
                                     float* __restrict__ refin) {
  int idx = blockIdx.x * 256 + threadIdx.x;
  if (idx >= 4 * 16 * H0 * W0) return;
  int r = idx / (16 * H0 * W0);
  int pi = idx % (16 * H0 * W0);
  float v = x[pi];
  refin[idx] = 1.f - log1pf(expm1f((1.f - v) * 8.7f) + c_deltas[r]) / 8.7f;
}

// patch embed; outputs channels-last (p*CE + ch), ch = d*16 + oc
__global__ void embed_kernel(const float* __restrict__ x, const float* __restrict__ refin,
                             const float* __restrict__ pew, const float* __restrict__ peb,
                             const float* __restrict__ perw, const float* __restrict__ perb,
                             _Float16* __restrict__ xh16, float* __restrict__ ref32,
                             _Float16* __restrict__ refh16) {
  int idx = blockIdx.x * 256 + threadIdx.x;
  if (idx >= 5 * NHW) return;
  int ch = idx & (CE - 1);
  int q = idx >> 7;            // s*HW + p
  int p = q % HW;
  int s = q / HW;
  int d = ch >> 4, oc = ch & 15;
  int oh = p / WPIX, ow = p % WPIX;
  const float* src = (s == 0) ? x : (refin + (size_t)(s - 1) * 16 * H0 * W0);
  const float* wgt = (s == 0) ? pew : perw;
  const float* bia = (s == 0) ? peb : perb;
  float acc = bia[oc];
#pragma unroll
  for (int ic = 0; ic < 2; ++ic)
#pragma unroll
    for (int kh = 0; kh < 2; ++kh)
#pragma unroll
      for (int kw = 0; kw < 2; ++kw)
        acc = fmaf(src[(size_t)(d * 2 + ic) * H0 * W0 + (2 * oh + kh) * W0 + (2 * ow + kw)],
                   wgt[((oc * 2 + ic) * 2 + kh) * 2 + kw], acc);
  size_t o = (size_t)p * CE + ch;
  if (s == 0) {
    xh16[o] = (_Float16)acc;
  } else {
    ref32[(size_t)(s - 1) * NHW + o] = acc;
    refh16[(size_t)(s - 1) * NHW + o] = (_Float16)acc;
  }
}

// ---------------------------------------------------------------------------
// Windowed cross attention: one 128-thread block (4 waves) per (2,4,4) window
// Activations channels-last: xh[(h*WPIX+w)*CE + d*C + c]
// ---------------------------------------------------------------------------

template <int C>
__global__ void __launch_bounds__(128)
attn_kernel(const _Float16* __restrict__ xh, const _Float16* __restrict__ refh,
            const _Float16* __restrict__ wq, const _Float16* __restrict__ wk,
            const _Float16* __restrict__ wv, const _Float16* __restrict__ wo,
            float* __restrict__ out) {
  constexpr int HD = C / 4;
  constexpr int CK = (C < 32) ? 32 : C;
  constexpr int NT = C / 16;
  constexpr int KSTEPS = CK / 32;
  constexpr int CH8 = C / 8;       // 16-byte chunks per token
  constexpr int KMAXC = (C < 32) ? C : 32;

  constexpr int SZ_XQ = 32 * CK * 2;
  constexpr int SZ_XKV = 128 * CK * 2;
  constexpr int SZ_HEAD = (SZ_XQ + SZ_XKV > 25600) ? (SZ_XQ + SZ_XKV) : 25600;
  constexpr int OFF_Q = SZ_HEAD;
  constexpr int OFF_K = OFF_Q + 32 * CK * 2;
  constexpr int OFF_V = OFF_K + 128 * CK * 2;
  constexpr int BYTES = OFF_V + 128 * CK * 2;   // C=64 -> 62464 B

  __shared__ __attribute__((aligned(16))) char smem[BYTES];
  _Float16* XQ  = (_Float16*)(smem);                 // 32 x CK tokens (dead after proj)
  _Float16* XKV = (_Float16*)(smem + SZ_XQ);         // 128 x CK tokens
  float*    Sf  = (float*)(smem);                    // 32 x 128 scores (aliases XQ/XKV)
  _Float16* Pf  = (_Float16*)(smem + 16384);         // 32 x 128 probs
  float*    Mred = (float*)(smem + 24576);           // 4 x 32 partial max
  float*    Rred = (float*)(smem + 25088);           // 4 x 32 partial sum
  _Float16* Qs  = (_Float16*)(smem + OFF_Q);         // 32 x CK; O overwrites per head
  _Float16* Ks  = (_Float16*)(smem + OFF_K);         // 128 x CK (token-major)
  _Float16* Vt  = (_Float16*)(smem + OFF_V);         // CK x 128 (transposed V)
  _Float16* Os  = Qs;

  const int tid = threadIdx.x;
  const int wave = tid >> 5;
  const int lane = tid & 31;
  const int l15 = lane & 15;
  const int kb8 = ((lane >> 4) & 1) * 8;
  const int kb16 = ((lane >> 4) & 1) * 16;

  const int wid = blockIdx.x;
  const int dw = wid / (32 * 40);
  const int rm = wid % (32 * 40);
  const int hw = rm / 40;
  const int ww = rm % 40;

  // zero-pad token LDS columns >= C (only when C < 32)
  if constexpr (C < 32) {
    for (int idx = tid; idx < 160 * 2; idx += 128) {   // 160 rows x 16 halves = two uint4 per row
      _Float16* base = (idx < 64) ? (XQ + (idx >> 1) * CK) : (XKV + ((idx - 64) >> 1) * CK);
      *(uint4*)(base + C + (idx & 1) * 8) = make_uint4(0, 0, 0, 0);
    }
  }
  // gather q tokens: 32 tokens x C contiguous halves, 16B chunks
  for (int idx = tid; idx < 32 * CH8; idx += 128) {
    int t = idx / CH8, cc = idx % CH8;
    int dz = t >> 4, hy = (t >> 2) & 3, wx = t & 3;
    int d = dw * 2 + dz, h = hw * 4 + hy, w = ww * 4 + wx;
    copy16_g2l(xh + (size_t)(h * WPIX + w) * CE + d * C + cc * 8,
               XQ + t * CK + cc * 8);
  }
  // gather kv tokens: 4 ref streams x 32 tokens
  for (int idx = tid; idx < 128 * CH8; idx += 128) {
    int kt = idx / CH8, cc = idx % CH8;
    int r = kt >> 5, t = kt & 31;
    int dz = t >> 4, hy = (t >> 2) & 3, wx = t & 3;
    int d = dw * 2 + dz, h = hw * 4 + hy, w = ww * 4 + wx;
    copy16_g2l(refh + ((size_t)r * HW + h * WPIX + w) * CE + d * C + cc * 8,
               XKV + kt * CK + cc * 8);
  }
  async_copies_wait();
  __syncthreads();

  // ---- Q projection (2*NT tiles over 4 waves) ----
  for (int t = wave; t < 2 * NT; t += 4) {
    int mt = t / NT, nt = t % NT;
    v8f acc = {0.f, 0.f, 0.f, 0.f, 0.f, 0.f, 0.f, 0.f};
#pragma unroll
    for (int ks = 0; ks < KSTEPS; ++ks) {
      v16h a = ld_a_fast(XQ + (mt * 16 + l15) * CK + ks * 32 + kb8);
      v16h b = ld_frag_b_pk(wq, ks * NT + nt);
      acc = WMMA_F16(a, b, acc);
    }
    st_frag_d_f16(Qs, CK, mt * 16, nt * 16, 16, acc);
  }
  // ---- K projection (8*NT tiles) ----
  for (int t = wave; t < 8 * NT; t += 4) {
    int mt = t / NT, nt = t % NT;
    v8f acc = {0.f, 0.f, 0.f, 0.f, 0.f, 0.f, 0.f, 0.f};
#pragma unroll
    for (int ks = 0; ks < KSTEPS; ++ks) {
      v16h a = ld_a_fast(XKV + (mt * 16 + l15) * CK + ks * 32 + kb8);
      v16h b = ld_frag_b_pk(wk, ks * NT + nt);
      acc = WMMA_F16(a, b, acc);
    }
    st_frag_d_f16(Ks, CK, mt * 16, nt * 16, 16, acc);
  }
  // ---- V projection (8*NT tiles), stored transposed ----
  for (int t = wave; t < 8 * NT; t += 4) {
    int mt = t / NT, nt = t % NT;
    v8f acc = {0.f, 0.f, 0.f, 0.f, 0.f, 0.f, 0.f, 0.f};
#pragma unroll
    for (int ks = 0; ks < KSTEPS; ++ks) {
      v16h a = ld_a_fast(XKV + (mt * 16 + l15) * CK + ks * 32 + kb8);
      v16h b = ld_frag_b_pk(wv, ks * NT + nt);
      acc = WMMA_F16(a, b, acc);
    }
    st_frag_d_f16_T(Vt, mt * 16, nt * 16, acc);
  }
  __syncthreads();

  const float scale = rsqrtf((float)HD);
  const int row = tid & 31;
  const int seg = tid >> 5;

  for (int h = 0; h < 4; ++h) {
    // S = Q_h * K_h^T  (2 x 8 tiles, head K padded to 32 via masking)
    for (int t = wave; t < 16; t += 4) {
      int mt = t >> 3, nt = t & 7;
      v8f z = {0.f, 0.f, 0.f, 0.f, 0.f, 0.f, 0.f, 0.f};
      v16h a = mask_a<HD>(ld_a_fast(Qs + h * HD + (mt * 16 + l15) * CK + kb8), kb8);
      v16h b = mask_a<HD>(ld_a_fast(Ks + h * HD + (nt * 16 + l15) * CK + kb8), kb8);
      v8f s = WMMA_F16(a, b, z);
      st_frag_d_f32(Sf, 128, mt * 16, nt * 16, s);
    }
    __syncthreads();
    // softmax: 32 rows x 4 column segments, all 128 threads
    {
      float* Sr = Sf + row * 128 + seg * 32;
      float mx = -1e30f;
      for (int k = 0; k < 32; ++k) mx = fmaxf(mx, Sr[k] * scale);
      Mred[seg * 32 + row] = mx;
      __syncthreads();
      float gm = fmaxf(fmaxf(Mred[row], Mred[32 + row]),
                       fmaxf(Mred[64 + row], Mred[96 + row]));
      float sum = 0.f;
      for (int k = 0; k < 32; ++k) {
        float e = __expf(Sr[k] * scale - gm);
        Sr[k] = e;
        sum += e;
      }
      Rred[seg * 32 + row] = sum;
      __syncthreads();
      float inv = 1.f / (Rred[row] + Rred[32 + row] + Rred[64 + row] + Rred[96 + row]);
      for (int k = 0; k < 32; ++k)
        Pf[row * 128 + seg * 32 + k] = (_Float16)(Sr[k] * inv);
    }
    __syncthreads();
    // O_h = P * V_h (V read from transposed tile; overwrites dead Q_h columns)
    for (int mt = wave; mt < 2; mt += 4) {
      int vr = h * HD + l15;             // V channel row in Vt
      if constexpr (HD < 16) { if (vr >= CK) vr = 0; }   // lanes beyond head: unused
      v8f acc = {0.f, 0.f, 0.f, 0.f, 0.f, 0.f, 0.f, 0.f};
#pragma unroll
      for (int ks = 0; ks < 4; ++ks) {
        v16h a = ld_a_fast(Pf + (mt * 16 + l15) * 128 + ks * 32 + kb8);
        v16h b = ld_a_fast(Vt + vr * 128 + ks * 32 + kb8);
        acc = WMMA_F16(a, b, acc);
      }
      st_frag_d_f16(Os + h * HD, CK, mt * 16, 0, HD, acc);
    }
    __syncthreads();
  }

  // Out = O * Wo, channels-last scatter: out[p*CE + d*C + n]
  for (int t = wave; t < 2 * NT; t += 4) {
    int mt = t / NT, nt = t % NT;
    v8f acc = {0.f, 0.f, 0.f, 0.f, 0.f, 0.f, 0.f, 0.f};
#pragma unroll
    for (int ks = 0; ks < KSTEPS; ++ks) {
      v16h a = mask_a<KMAXC>(ld_a_fast(Os + (mt * 16 + l15) * CK + ks * 32 + kb8), kb8);
      v16h b = ld_frag_b_pk(wo, ks * NT + nt);
      acc = WMMA_F16(a, b, acc);
    }
    int n = nt * 16 + l15;
#pragma unroll
    for (int r = 0; r < 8; ++r) {
      int m = mt * 16 + r + ((lane >> 4) << 3);
      int dz = m >> 4, hy = (m >> 2) & 3, wx = m & 3;
      int d = dw * 2 + dz, hh = hw * 4 + hy, w2 = ww * 4 + wx;
      out[(size_t)(hh * WPIX + w2) * CE + d * C + n] = acc[r];
    }
  }
}

// ---------------------------------------------------------------------------
// Whole-tensor LayerNorm (two-stage reduce) + grouped 1x1 conv + ReLU
// ---------------------------------------------------------------------------

__global__ void reduce_partial_kernel(const float* __restrict__ in,
                                      float* __restrict__ part) {
  __shared__ float s1[256], s2[256];
  int s = blockIdx.y;
  const float* p = in + (size_t)s * NHW;
  float a = 0.f, b = 0.f;
  for (size_t i = (size_t)blockIdx.x * 256 + threadIdx.x; i < NHW;
       i += (size_t)gridDim.x * 256) {
    __builtin_prefetch(&p[i + (size_t)gridDim.x * 256], 0, 1);
    float v = p[i];
    a += v;
    b += v * v;
  }
  s1[threadIdx.x] = a;
  s2[threadIdx.x] = b;
  __syncthreads();
  for (int o = 128; o > 0; o >>= 1) {
    if (threadIdx.x < o) {
      s1[threadIdx.x] += s1[threadIdx.x + o];
      s2[threadIdx.x] += s2[threadIdx.x + o];
    }
    __syncthreads();
  }
  if (threadIdx.x == 0) {
    part[(s * 256 + blockIdx.x) * 2 + 0] = s1[0];
    part[(s * 256 + blockIdx.x) * 2 + 1] = s2[0];
  }
}

__global__ void ln_finalize_kernel(const float* __restrict__ part,
                                   float* __restrict__ stats) {
  __shared__ float s1[256], s2[256];
  int s = blockIdx.x;
  s1[threadIdx.x] = part[(s * 256 + threadIdx.x) * 2 + 0];
  s2[threadIdx.x] = part[(s * 256 + threadIdx.x) * 2 + 1];
  __syncthreads();
  for (int o = 128; o > 0; o >>= 1) {
    if (threadIdx.x < o) {
      s1[threadIdx.x] += s1[threadIdx.x + o];
      s2[threadIdx.x] += s2[threadIdx.x + o];
    }
    __syncthreads();
  }
  if (threadIdx.x == 0) {
    float m = s1[0] / (float)NHW;
    float v = s2[0] / (float)NHW - m * m;
    stats[s * 2 + 0] = m;
    stats[s * 2 + 1] = rsqrtf(v + 1e-5f);
  }
}

// block = 256 threads = 2 pixels x 128 channels; inputs/outputs channels-last.
// chw32 != 0 -> write out32 channel-major (final d_out layout).
__global__ void ln_conv_kernel(const float* __restrict__ in,
                               const float* __restrict__ lnw, const float* __restrict__ lnb,
                               const float* __restrict__ cw, const float* __restrict__ cb,
                               const float* __restrict__ stats,
                               float* __restrict__ out32, _Float16* __restrict__ out16,
                               int gs, int chw32) {
  __shared__ float nrm[2][CE];
  int s = blockIdx.y;
  const float* pin = in + (size_t)s * NHW;
  _Float16* p16 = out16 + (size_t)s * NHW;
  float* p32 = out32 ? (out32 + (size_t)s * NHW) : nullptr;
  float mean = stats[s * 2 + 0], istd = stats[s * 2 + 1];
  int local = threadIdx.x >> 7;         // pixel within block
  int ch = threadIdx.x & (CE - 1);
  int p = blockIdx.x * 2 + local;
  // normalize (whole-tensor LN with per-element affine)
  size_t e = (size_t)ch * HW + p;       // param layout is channel-major
  nrm[local][ch] = fmaf((pin[(size_t)p * CE + ch] - mean) * istd, lnw[e], lnb[e]);
  __syncthreads();
  // grouped 1x1 conv + ReLU, oc == ch
  int gb = (ch / gs) * gs;
  float acc = cb[ch];
  for (int j = 0; j < gs; ++j) acc = fmaf(cw[ch * gs + j], nrm[local][gb + j], acc);
  float y = fmaxf(acc, 0.f);
  if (p32) p32[chw32 ? e : ((size_t)p * CE + ch)] = y;
  p16[(size_t)p * CE + ch] = (_Float16)y;
}

// ---------------------------------------------------------------------------
// Host launcher
// ---------------------------------------------------------------------------

extern "C" void kernel_launch(void* const* d_in, const int* in_sizes, int n_in,
                              void* d_out, int out_size, void* d_ws, size_t ws_size,
                              hipStream_t stream) {
  (void)in_sizes; (void)n_in; (void)out_size; (void)ws_size;

  const float* x    = (const float*)d_in[0];
  const float* pew  = (const float*)d_in[1];
  const float* peb  = (const float*)d_in[2];
  const float* perw = (const float*)d_in[3];
  const float* perb = (const float*)d_in[4];
  const float* lnw[3]  = {(const float*)d_in[5],  (const float*)d_in[6],  (const float*)d_in[7]};
  const float* lnb[3]  = {(const float*)d_in[8],  (const float*)d_in[9],  (const float*)d_in[10]};
  const float* cw[3]   = {(const float*)d_in[11], (const float*)d_in[12], (const float*)d_in[13]};
  const float* cb[3]   = {(const float*)d_in[14], (const float*)d_in[15], (const float*)d_in[16]};
  const float* lnrw[2] = {(const float*)d_in[17], (const float*)d_in[18]};
  const float* lnrb[2] = {(const float*)d_in[19], (const float*)d_in[20]};
  const float* cwr[2]  = {(const float*)d_in[21], (const float*)d_in[22]};
  const float* cbr[2]  = {(const float*)d_in[23], (const float*)d_in[24]};
  const float* wqf[3]  = {(const float*)d_in[25], (const float*)d_in[26], (const float*)d_in[27]};
  const float* wkf[3]  = {(const float*)d_in[28], (const float*)d_in[29], (const float*)d_in[30]};
  const float* wvf[3]  = {(const float*)d_in[31], (const float*)d_in[32], (const float*)d_in[33]};
  const float* wof[3]  = {(const float*)d_in[34], (const float*)d_in[35], (const float*)d_in[36]};

  char* ws = (char*)d_ws;
  size_t cur = 0;
  auto take = [&](size_t bytes) -> void* {
    void* p = ws + cur;
    cur = (cur + bytes + 255) & ~(size_t)255;
    return p;
  };

  // packed weight sizes per matrix: (CK/32)*(C/16)*512 halves
  const int Cs[3] = {16, 32, 64};
  const size_t psz[3] = {512, 1024, 4096};
  const size_t pb[3] = {0, 4 * 512, 4 * 512 + 4 * 1024};   // layer bases (halves)

  _Float16* wt16   = (_Float16*)take((size_t)(4 * 512 + 4 * 1024 + 4 * 4096) * 2);
  _Float16* xh16   = (_Float16*)take((size_t)NHW * 2);
  _Float16* refh16 = (_Float16*)take((size_t)4 * NHW * 2);
  float* refA      = (float*)take((size_t)4 * NHW * 4);
  float* refB      = (float*)take((size_t)4 * NHW * 4);
  float* attn32    = (float*)take((size_t)NHW * 4);
  float* partX     = (float*)take((size_t)256 * 2 * 4);
  float* partR     = (float*)take((size_t)4 * 256 * 2 * 4);
  float* statsX    = (float*)take((size_t)2 * 4);
  float* statsR    = (float*)take((size_t)8 * 4);
  float* refin     = refB;  // transform scratch aliases refB (dead until layer 0 conv)

  // pack q/k/v/o weights into WMMA B-fragment order (f16, K zero-padded)
  for (int i = 0; i < 3; ++i) {
    int CKp = (Cs[i] < 32) ? 32 : Cs[i];
    int gb = ((int)psz[i] + 255) / 256;
    pack_w_kernel<<<gb, 256, 0, stream>>>(wqf[i], wt16 + pb[i] + 0 * psz[i], Cs[i], CKp);
    pack_w_kernel<<<gb, 256, 0, stream>>>(wkf[i], wt16 + pb[i] + 1 * psz[i], Cs[i], CKp);
    pack_w_kernel<<<gb, 256, 0, stream>>>(wvf[i], wt16 + pb[i] + 2 * psz[i], Cs[i], CKp);
    pack_w_kernel<<<gb, 256, 0, stream>>>(wof[i], wt16 + pb[i] + 3 * psz[i], Cs[i], CKp);
  }

  // transform + patch embed
  ref_transform_kernel<<<(4 * 16 * H0 * W0) / 256, 256, 0, stream>>>(x, refin);
  embed_kernel<<<(5 * NHW) / 256, 256, 0, stream>>>(x, refin, pew, peb, perw, perb,
                                                    xh16, refA, refh16);

  const int gsArr[3] = {32, 64, 128};
  float* refCur = refA;
  float* refNxt = refB;

  for (int i = 0; i < 3; ++i) {
    const _Float16* q16 = wt16 + pb[i] + 0 * psz[i];
    const _Float16* k16 = wt16 + pb[i] + 1 * psz[i];
    const _Float16* v16 = wt16 + pb[i] + 2 * psz[i];
    const _Float16* o16 = wt16 + pb[i] + 3 * psz[i];

    if (i == 0)
      attn_kernel<16><<<5120, 128, 0, stream>>>(xh16, refh16, q16, k16, v16, o16, attn32);
    else if (i == 1)
      attn_kernel<32><<<2560, 128, 0, stream>>>(xh16, refh16, q16, k16, v16, o16, attn32);
    else
      attn_kernel<64><<<1280, 128, 0, stream>>>(xh16, refh16, q16, k16, v16, o16, attn32);

    // x stream: LN stats + fused LN/conv/ReLU
    reduce_partial_kernel<<<dim3(256, 1), 256, 0, stream>>>(attn32, partX);
    ln_finalize_kernel<<<1, 256, 0, stream>>>(partX, statsX);
    float* o32 = (i == 2) ? (float*)d_out : nullptr;
    ln_conv_kernel<<<dim3(HW / 2, 1), 256, 0, stream>>>(
        attn32, lnw[i], lnb[i], cw[i], cb[i], statsX, o32, xh16, gsArr[i], 1);

    // ref streams (layers 0,1 only)
    if (i < 2) {
      reduce_partial_kernel<<<dim3(256, 4), 256, 0, stream>>>(refCur, partR);
      ln_finalize_kernel<<<4, 256, 0, stream>>>(partR, statsR);
      ln_conv_kernel<<<dim3(HW / 2, 4), 256, 0, stream>>>(
          refCur, lnrw[i], lnrb[i], cwr[i], cbr[i], statsR, refNxt, refh16, gsArr[i], 0);
      float* t = refCur; refCur = refNxt; refNxt = t;
    }
  }
}
